// LipNet_Pinyin_19207093747843
// MI455X (gfx1250) — matrix operand order repaired and checked by reading, not compiled, not run on hardware
//
#include <hip/hip_runtime.h>
#include <math.h>

// ---------------------------------------------------------------------------
// CDNA5 (gfx1250) LipNet-Pinyin forward.  bf16 WMMA everywhere; f32 state.
// GEMMs: 4-wave workgroups, A panel async-staged to LDS (double buffered,
// K % 128 == 0 so the inner loop is a compile-time 4x WMMA unroll),
// B streamed from global.  Recurrence: persistent 32-wave WMMA kernel.
// ---------------------------------------------------------------------------

typedef __attribute__((ext_vector_type(16))) __bf16        v16bf;
typedef __attribute__((ext_vector_type(8)))  float         v8f;
typedef __attribute__((ext_vector_type(4)))  unsigned int  u32x4;

union V16U { v16bf v; u32x4 q[2]; };

// A-operand (16-bit 16x32): lane holds row M=l&15; contiguous K run
// [lh*16, lh*16+16) (two 16B runs).
__device__ __forceinline__ v16bf ldA16(const __bf16* base, int k0, int lh) {
  V16U u;
  u.q[0] = *reinterpret_cast<const u32x4*>(base + k0 + lh * 16);
  u.q[1] = *reinterpret_cast<const u32x4*>(base + k0 + lh * 16 + 8);
  return u.v;
}
// B-operand (32x16): lane holds column N=l&15; contiguous K run [lh*16, lh*16+16)
__device__ __forceinline__ v16bf ldB16(const __bf16* base, int k0, int lh) {
  V16U u;
  u.q[0] = *reinterpret_cast<const u32x4*>(base + k0 + lh * 16);
  u.q[1] = *reinterpret_cast<const u32x4*>(base + k0 + lh * 16 + 8);
  return u.v;
}
__device__ __forceinline__ v8f wmma_bf(v16bf a, v16bf b, v8f c) {
  return __builtin_amdgcn_wmma_f32_16x16x32_bf16(false, a, false, b,
                                                 (short)0, c, false, false);
}
__device__ __forceinline__ float sigm(float x) { return 1.0f / (1.0f + __expf(-x)); }

// ---------------------------------------------------------------------------
// Cooperative GEMM: C[M,N] = act( A_bf16[M,K] * W_bf16[N,K]^T + bias ).
// Block = 4 waves sharing one 16-row A panel (async-staged to LDS, 16x128
// double-buffered chunks); each wave owns one 16x16 output tile.
// Requires Npad % 64 == 0, M % 16 == 0, K % 128 == 0.
// ---------------------------------------------------------------------------
__global__ void __launch_bounds__(128)
gemm_bf16_kernel(const __bf16* __restrict__ A, int lda,
                 const __bf16* __restrict__ W, int ldw,
                 float* __restrict__ C, int ldc,
                 __bf16* __restrict__ Cb, int ldcb,
                 const float* __restrict__ bias,
                 int K, int Nreal, int act) {
  __shared__ __align__(16) __bf16 sA[2][16][128];
  const int tid = threadIdx.x;
  const int wv  = tid >> 5;
  const int l   = tid & 31;
  const int l15 = l & 15, lh = l >> 4;
  const long m0 = (long)blockIdx.x * 16;
  const int  n  = blockIdx.y * 64 + wv * 16 + l15;

  const __bf16* Abase = A + m0 * (long)lda;

  // Async-stage one 16x128 bf16 panel into sA[buf] (ASYNCcnt-tracked).
  // 256 x 16B segments; 128 threads -> fully unrolled 2 per thread.
  auto stage = [&](int buf, int kc) {
#pragma unroll
    for (int rep = 0; rep < 2; ++rep) {
      const int s = tid + rep * 128;
      const int row = s >> 4, col = s & 15;
      const __bf16* g = Abase + (long)row * lda + kc + col * 8;
      unsigned lds = (unsigned)(size_t)(&sA[buf][row][col * 8]);
      asm volatile("global_load_async_to_lds_b128 %0, %1, off"
                   :: "v"(lds), "v"(g) : "memory");
    }
  };

  float bv = 0.f;
  if (bias && n < Nreal) bv = bias[n];
  v8f acc;
#pragma unroll
  for (int i = 0; i < 8; ++i) acc[i] = bv;

  const __bf16* wrow = W + (long)n * ldw;

  stage(0, 0);
  asm volatile("s_wait_asynccnt 0x0" ::: "memory");
  __syncthreads();

  int cur = 0;
  for (int kc = 0; kc < K; kc += 128) {
    const int nkc = kc + 128;
    if (nkc < K) stage(cur ^ 1, nkc);

    const __bf16* arow = &sA[cur][l15][0];
#pragma unroll
    for (int k0 = 0; k0 < 128; k0 += 32) {
      v16bf a = ldA16(arow, k0, lh);              // ds_load_b128 from LDS
      v16bf b = ldB16(wrow, kc + k0, lh);         // global_load_b128
      acc = wmma_bf(a, b, acc);
    }
    if (nkc < K) {
      asm volatile("s_wait_asynccnt 0x0" ::: "memory");
      __syncthreads();
    }
    cur ^= 1;
  }

  if (n < Nreal) {
#pragma unroll
    for (int v = 0; v < 8; ++v) {
      long row = m0 + v + 8 * lh;
      float val = acc[v];
      if (act == 1) val = fmaxf(val, 0.f);
      C[row * ldc + n] = val;
      if (Cb) Cb[row * ldcb + n] = (__bf16)val;
    }
  }
}

// ---------------------------------------------------------------------------
// Persistent GRU recurrence: one 32-wave workgroup scans T steps.
//   gh[16,3072] = h_bf16[16,1024] * Whh[3072,1024]^T + bhh  via WMMA,
// wave w owns h-columns [32w,32w+32) -> its 6 accumulators feed gate math
// directly in registers.  h kept in LDS (f32 master + bf16 operand copy).
// ---------------------------------------------------------------------------
__global__ void __launch_bounds__(1024)
gru_scan_kernel(const float* __restrict__ gi,    // [T][16][3072]  (x@Wih^T+bih)
                const __bf16* __restrict__ Whh,  // [3072][1024]
                const float* __restrict__ bhh,   // [3072]
                float* __restrict__ ys,          // [T][16][1024] or null
                __bf16* __restrict__ xsn,        // [T][16][xstride] or null
                int xstride, int xoff,
                float* __restrict__ hT,          // [16][1024]
                int T, int rev) {
  extern __shared__ char smem[];
  __bf16* hbf = (__bf16*)smem;                                // 32 KB
  float*  hf  = (float*)(smem + 16 * 1024 * sizeof(__bf16));  // 64 KB

  const int tid = threadIdx.x;
  const int w = tid >> 5, l = tid & 31;
  const int l15 = l & 15, lh = l >> 4;
  const int c0 = w * 32;

  for (int i = tid; i < 16 * 1024; i += 1024) { hf[i] = 0.f; hbf[i] = (__bf16)0.f; }
  __syncthreads();

  float bb[6];
#pragma unroll
  for (int g = 0; g < 3; ++g)
#pragma unroll
    for (int s = 0; s < 2; ++s)
      bb[g * 2 + s] = bhh[g * 1024 + c0 + s * 16 + l15];

  for (int step = 0; step < T; ++step) {
    const int tt = rev ? (T - 1 - step) : step;
    const float* giT = gi + (size_t)tt * (16 * 3072);

    v8f acc[6];
#pragma unroll
    for (int j = 0; j < 6; ++j)
#pragma unroll
      for (int i = 0; i < 8; ++i) acc[j][i] = bb[j];

    const __bf16* arow = hbf + l15 * 1024;
#pragma unroll 2
    for (int k0 = 0; k0 < 1024; k0 += 32) {
      v16bf a = ldA16(arow, k0, lh);
#pragma unroll
      for (int g = 0; g < 3; ++g)
#pragma unroll
        for (int s = 0; s < 2; ++s) {
          const __bf16* wrow = Whh + (size_t)(g * 1024 + c0 + s * 16 + l15) * 1024;
          v16bf b = ldB16(wrow, k0, lh);
          acc[g * 2 + s] = wmma_bf(a, b, acc[g * 2 + s]);
        }
    }
    if (step + 1 < T) {  // warm L2/WGP$ with next step's gate-input slab
      const int tn = rev ? (T - 2 - step) : (step + 1);
      __builtin_prefetch(gi + (size_t)tn * (16 * 3072) + tid * 48, 0, 0);
    }
    __syncthreads();  // all WMMA reads of hbf complete before state update

#pragma unroll
    for (int s = 0; s < 2; ++s) {
      const int c = c0 + s * 16 + l15;
#pragma unroll
      for (int v = 0; v < 8; ++v) {
        const int m = v + 8 * lh;
        const float* gim = giT + m * 3072;
        const float r  = sigm(gim[c] + acc[0 * 2 + s][v]);
        const float z  = sigm(gim[1024 + c] + acc[1 * 2 + s][v]);
        const float nn = tanhf(gim[2048 + c] + r * acc[2 * 2 + s][v]);
        const float hn = (1.f - z) * nn + z * hf[m * 1024 + c];
        hf[m * 1024 + c]  = hn;
        hbf[m * 1024 + c] = (__bf16)hn;
        if (ys)  ys[(size_t)tt * 16384 + m * 1024 + c] = hn;
        if (xsn) xsn[(size_t)tt * 16 * xstride + m * xstride + xoff + c] = (__bf16)hn;
      }
    }
    __syncthreads();
  }
  for (int i = tid; i < 16 * 1024; i += 1024) hT[i] = hf[i];
}

// ---------------------------------------------------------------------------
// Elementwise / conversion helpers
// ---------------------------------------------------------------------------
__global__ void pack_encin_kernel(const float* __restrict__ x,
                                  const float* __restrict__ border,
                                  __bf16* __restrict__ dst) {
  int i = blockIdx.x * 256 + threadIdx.x;
  if (i >= 200 * 16 * 640) return;
  int t = i / (16 * 640), r = i % (16 * 640);
  int b = r / 640, k = r % 640;
  float v = 0.f;
  if (k < 512) v = x[((size_t)b * 200 + t) * 512 + k];
  else if (k == 512) v = border[b * 200 + t];
  dst[i] = (__bf16)v;
}

__global__ void conv_nt_kernel(const float* __restrict__ src, __bf16* __restrict__ dst,
                               int N, int K, int Kp) {
  long i = (long)blockIdx.x * 256 + threadIdx.x;
  if (i >= (long)N * Kp) return;
  int n = (int)(i / Kp), k = (int)(i % Kp);
  dst[i] = (k < K) ? (__bf16)src[(size_t)n * K + k] : (__bf16)0.f;
}

__global__ void conv_t_kernel(const float* __restrict__ src, int srcld, int row0,
                              __bf16* __restrict__ dst, int N, int K, int Np) {
  long i = (long)blockIdx.x * 256 + threadIdx.x;
  if (i >= (long)Np * K) return;
  int n = (int)(i / K), k = (int)(i % K);
  dst[i] = (n < N) ? (__bf16)src[(size_t)(row0 + k) * srcld + n] : (__bf16)0.f;
}

__global__ void f32_to_bf16_kernel(const float* __restrict__ s, __bf16* __restrict__ d, int n) {
  int i = blockIdx.x * 256 + threadIdx.x;
  if (i < n) d[i] = (__bf16)s[i];
}
__global__ void zero_f32_kernel(float* p, int n) {
  int i = blockIdx.x * 256 + threadIdx.x;
  if (i < n) p[i] = 0.f;
}

// LayerNorm over 2048 (concat fwd/bwd), writes bf16 GEMM operand.
__global__ void ln_enc_kernel(const float* __restrict__ ysf, const float* __restrict__ ysb,
                              const float* __restrict__ g, const float* __restrict__ be,
                              __bf16* __restrict__ A) {
  const int i = blockIdx.x;            // b*200 + t
  const int b = i / 200, t = i % 200;
  const int tid = threadIdx.x;
  __shared__ float r1[256], r2[256];
  float s = 0.f, s2 = 0.f;
  for (int j = tid; j < 2048; j += 256) {
    float v = (j < 1024) ? ysf[((size_t)t * 16 + b) * 1024 + j]
                         : ysb[((size_t)t * 16 + b) * 1024 + (j - 1024)];
    s += v; s2 += v * v;
  }
  r1[tid] = s; r2[tid] = s2; __syncthreads();
  for (int o = 128; o > 0; o >>= 1) {
    if (tid < o) { r1[tid] += r1[tid + o]; r2[tid] += r2[tid + o]; }
    __syncthreads();
  }
  const float mu = r1[0] * (1.f / 2048.f);
  const float var = r2[0] * (1.f / 2048.f) - mu * mu;
  const float rstd = rsqrtf(var + 1e-5f);
  for (int j = tid; j < 2048; j += 256) {
    float v = (j < 1024) ? ysf[((size_t)t * 16 + b) * 1024 + j]
                         : ysb[((size_t)t * 16 + b) * 1024 + (j - 1024)];
    A[(size_t)i * 2048 + j] = (__bf16)((v - mu) * rstd * g[j] + be[j]);
  }
}

__global__ void ln_dec_kernel(const float* __restrict__ hfst, const float* __restrict__ hbst,
                              const float* __restrict__ g, const float* __restrict__ be,
                              __bf16* __restrict__ A /* [16][6144], cols 0..2047 */) {
  const int b = blockIdx.x, tid = threadIdx.x;
  __shared__ float r1[256], r2[256];
  float s = 0.f, s2 = 0.f;
  for (int j = tid; j < 2048; j += 256) {
    float v = (j < 1024) ? hfst[b * 1024 + j] : hbst[b * 1024 + (j - 1024)];
    s += v; s2 += v * v;
  }
  r1[tid] = s; r2[tid] = s2; __syncthreads();
  for (int o = 128; o > 0; o >>= 1) {
    if (tid < o) { r1[tid] += r1[tid + o]; r2[tid] += r2[tid + o]; }
    __syncthreads();
  }
  const float mu = r1[0] * (1.f / 2048.f);
  const float rstd = rsqrtf(r2[0] * (1.f / 2048.f) - mu * mu + 1e-5f);
  for (int j = tid; j < 2048; j += 256) {
    float v = (j < 1024) ? hfst[b * 1024 + j] : hbst[b * 1024 + (j - 1024)];
    A[(size_t)b * 6144 + j] = (__bf16)((v - mu) * rstd * g[j] + be[j]);
  }
}

// Fused windowed attention: score -> masked softmax -> context; writes context
// f32 + bf16 into the decoder-GRU and decoder-linear operand rows.
__global__ void attn_step_kernel(const float* __restrict__ Epre,   // [16,200,1024]
                                 const float* __restrict__ Dproj,  // [16,1024]
                                 const float* __restrict__ vvec, const float* __restrict__ vb,
                                 const float* __restrict__ enc_out,// [16,200,2048]
                                 const int* __restrict__ st, const int* __restrict__ ed,
                                 float* __restrict__ cbuf, __bf16* __restrict__ gin,
                                 __bf16* __restrict__ adl) {
  const int b = blockIdx.x, tid = threadIdx.x;
  const int w = tid >> 5, l = tid & 31;
  __shared__ float sc[200];
  __shared__ float red[256];
  for (int t = w; t < 200; t += 8) {
    float dot = 0.f;
    const float* ep = Epre + ((size_t)b * 200 + t) * 1024;
    const float* dp = Dproj + b * 1024;
    for (int k = l; k < 1024; k += 32) dot += tanhf(ep[k] + dp[k]) * vvec[k];
#pragma unroll
    for (int o = 16; o > 0; o >>= 1) dot += __shfl_down(dot, o, 32);
    if (l == 0) sc[t] = dot + vb[0];
  }
  __syncthreads();
  const int s0 = st[b], e0 = ed[b];
  float sv = -3.0e38f;
  if (tid < 200) { sv = sc[tid]; if (tid < s0 || tid >= e0) sv = -3.0e38f; }
  red[tid] = sv; __syncthreads();
  for (int o = 128; o > 0; o >>= 1) { if (tid < o) red[tid] = fmaxf(red[tid], red[tid + o]); __syncthreads(); }
  const float mx = red[0]; __syncthreads();
  float e = (tid < 200) ? __expf(sv - mx) : 0.f;
  red[tid] = e; __syncthreads();
  for (int o = 128; o > 0; o >>= 1) { if (tid < o) red[tid] += red[tid + o]; __syncthreads(); }
  const float inv = 1.f / red[0]; __syncthreads();
  if (tid < 200) sc[tid] = e * inv;
  __syncthreads();
  for (int d = tid; d < 2048; d += 256) {
    float a = 0.f;
    for (int t = 0; t < 200; ++t) a += sc[t] * enc_out[((size_t)b * 200 + t) * 2048 + d];
    cbuf[b * 2048 + d] = a;
    __bf16 ab = (__bf16)a;
    gin[(size_t)b * 4096 + 2048 + d] = ab;
    adl[(size_t)b * 6144 + 2048 + d] = ab;
  }
}

__global__ void gru_combine_kernel(const float* __restrict__ gi, const float* __restrict__ gh,
                                   float* __restrict__ h, __bf16* __restrict__ hb) {
  int i = blockIdx.x * 256 + threadIdx.x;
  if (i >= 16 * 1024) return;
  int m = i >> 10, c = i & 1023;
  const float* gim = gi + m * 3072;
  const float* ghm = gh + m * 3072;
  float r  = sigm(gim[c] + ghm[c]);
  float z  = sigm(gim[1024 + c] + ghm[1024 + c]);
  float nn = tanhf(gim[2048 + c] + r * ghm[2048 + c]);
  float hn = (1.f - z) * nn + z * h[i];
  h[i] = hn; hb[i] = (__bf16)hn;
}

__global__ void emb_init_kernel(const int* __restrict__ tgt, const float* __restrict__ emb,
                                __bf16* __restrict__ decin, __bf16* __restrict__ gin,
                                __bf16* __restrict__ adl) {
  int i = blockIdx.x * 256 + threadIdx.x;
  if (i >= 16 * 2048) return;
  int b = i >> 11, d = i & 2047;
  int row = tgt[b * 40];
  __bf16 v = (__bf16)emb[(size_t)row * 2048 + d];
  decin[i] = v; gin[(size_t)b * 4096 + d] = v; adl[(size_t)b * 6144 + d] = v;
}

__global__ void dec_post_kernel(const float* __restrict__ out, float* __restrict__ osum,
                                __bf16* __restrict__ decin, __bf16* __restrict__ gin,
                                __bf16* __restrict__ adl) {
  int i = blockIdx.x * 256 + threadIdx.x;
  if (i >= 16 * 2048) return;
  int b = i >> 11, d = i & 2047;
  float o = out[i];
  osum[i] += o;
  __bf16 ob = (__bf16)o;
  decin[i] = ob; gin[(size_t)b * 4096 + d] = ob; adl[(size_t)b * 6144 + d] = ob;
}

__global__ void mean_kernel(const float* __restrict__ osum, __bf16* __restrict__ A) {
  int i = blockIdx.x * 256 + threadIdx.x;
  if (i < 16 * 2048) A[i] = (__bf16)(osum[i] * (1.f / 40.f));
}

// ---------------------------------------------------------------------------
// Host orchestration
// ---------------------------------------------------------------------------
static inline void gemm(hipStream_t s, const __bf16* A, int lda, const __bf16* W, int ldw,
                        float* C, int ldc, __bf16* Cb, int ldcb, const float* bias,
                        int M, int K, int Nreal, int Npad, int act) {
  dim3 g(M / 16, Npad / 64);   // 4 waves per block, one 16x16 tile each
  gemm_bf16_kernel<<<g, 128, 0, s>>>(A, lda, W, ldw, C, ldc, Cb, ldcb, bias, K, Nreal, act);
}
#define EW(kern, n, ...) kern<<<((n) + 255) / 256, 256, 0, stream>>>(__VA_ARGS__)

extern "C" void kernel_launch(void* const* d_in, const int* in_sizes, int n_in,
                              void* d_out, int out_size, void* d_ws, size_t ws_size,
                              hipStream_t stream) {
  (void)in_sizes; (void)n_in; (void)out_size; (void)ws_size;
  const float* X    = (const float*)d_in[0];
  const float* BORD = (const float*)d_in[1];
  const int*   TGT  = (const int*)d_in[2];
  const int*   SST  = (const int*)d_in[3];
  const int*   SED  = (const int*)d_in[4];
  // encoder GRUs: l0f, l0b, l1f, l1b  (Wih, Whh, bih, bhh each)
  const float *W_ih[4], *W_hh[4], *B_ih[4], *B_hh[4];
  for (int j = 0; j < 4; ++j) {
    W_ih[j] = (const float*)d_in[5 + 4 * j + 0];
    W_hh[j] = (const float*)d_in[5 + 4 * j + 1];
    B_ih[j] = (const float*)d_in[5 + 4 * j + 2];
    B_hh[j] = (const float*)d_in[5 + 4 * j + 3];
  }
  const float* ENC_LN_G = (const float*)d_in[21];
  const float* ENC_LN_B = (const float*)d_in[22];
  const float* ENC_LIN_W = (const float*)d_in[23];
  const float* ENC_LIN_B = (const float*)d_in[24];
  const float* ATTN_W = (const float*)d_in[25];
  const float* ATTN_B = (const float*)d_in[26];
  const float* ATTN_V = (const float*)d_in[27];
  const float* ATTN_VB = (const float*)d_in[28];
  const float *DW_ih[2], *DW_hh[2], *DB_ih[2], *DB_hh[2];
  for (int j = 0; j < 2; ++j) {
    DW_ih[j] = (const float*)d_in[29 + 4 * j + 0];
    DW_hh[j] = (const float*)d_in[29 + 4 * j + 1];
    DB_ih[j] = (const float*)d_in[29 + 4 * j + 2];
    DB_hh[j] = (const float*)d_in[29 + 4 * j + 3];
  }
  const float* DEC_LN_G = (const float*)d_in[37];
  const float* DEC_LN_B = (const float*)d_in[38];
  const float* DEC_LIN_W = (const float*)d_in[39];
  const float* DEC_LIN_B = (const float*)d_in[40];
  const float* EMB = (const float*)d_in[41];
  const float* CLS_W = (const float*)d_in[42];
  const float* CLS_B = (const float*)d_in[43];
  float* OUT = (float*)d_out;

  // ---- workspace carve ----
  char* ws = (char*)d_ws;
  size_t off = 0;
  auto carve = [&](size_t bytes) -> char* {
    off = (off + 255) & ~(size_t)255;
    char* p = ws + off; off += bytes; return p;
  };
  __bf16* Wc_ih0[2]; __bf16* Wc_hh[4]; __bf16* Wc_ih1[2];
  Wc_ih0[0] = (__bf16*)carve(3072UL * 640 * 2);
  Wc_ih0[1] = (__bf16*)carve(3072UL * 640 * 2);
  for (int j = 0; j < 4; ++j) Wc_hh[j] = (__bf16*)carve(3072UL * 1024 * 2);
  Wc_ih1[0] = (__bf16*)carve(3072UL * 2048 * 2);
  Wc_ih1[1] = (__bf16*)carve(3072UL * 2048 * 2);
  __bf16* WencL = (__bf16*)carve(2048UL * 2048 * 2);
  __bf16* WattE = (__bf16*)carve(1024UL * 2048 * 2);
  __bf16* WattD = (__bf16*)carve(1024UL * 2048 * 2);
  __bf16* Wd_ih[2]; __bf16* Wd_hh[2];
  for (int j = 0; j < 2; ++j) Wd_ih[j] = (__bf16*)carve(3072UL * 4096 * 2);
  for (int j = 0; j < 2; ++j) Wd_hh[j] = (__bf16*)carve(3072UL * 1024 * 2);
  __bf16* WdecL = (__bf16*)carve(2048UL * 6144 * 2);
  __bf16* Wcls  = (__bf16*)carve(1024UL * 2048 * 2);

  __bf16* A_encin = (__bf16*)carve(200UL * 16 * 640 * 2);
  float*  giA = (float*)carve(200UL * 16 * 3072 * 4);
  float*  giB = (float*)carve(200UL * 16 * 3072 * 4);
  __bf16* xs1 = (__bf16*)carve(200UL * 16 * 2048 * 2);
  float*  ys1f = (float*)carve(200UL * 16 * 1024 * 4);
  float*  ys1b = (float*)carve(200UL * 16 * 1024 * 4);
  __bf16* A_enc = (__bf16*)carve(3200UL * 2048 * 2);
  float*  enc_out = (float*)carve(3200UL * 2048 * 4);
  __bf16* A_encout = (__bf16*)carve(3200UL * 2048 * 2);
  float*  Epre = (float*)carve(3200UL * 1024 * 4);
  float*  hTf = (float*)carve(16UL * 1024 * 4);
  float*  hTb = (float*)carve(16UL * 1024 * 4);
  __bf16* hbf_f = (__bf16*)carve(16UL * 1024 * 2);
  __bf16* hbf_b = (__bf16*)carve(16UL * 1024 * 2);
  __bf16* decin = (__bf16*)carve(16UL * 2048 * 2);
  float*  Dproj = (float*)carve(16UL * 1024 * 4);
  __bf16* gin = (__bf16*)carve(16UL * 4096 * 2);
  float*  cbuf = (float*)carve(16UL * 2048 * 4);
  float*  gidf = (float*)carve(16UL * 3072 * 4);
  float*  gidb = (float*)carve(16UL * 3072 * 4);
  float*  ghdf = (float*)carve(16UL * 3072 * 4);
  float*  ghdb = (float*)carve(16UL * 3072 * 4);
  __bf16* A_dl = (__bf16*)carve(16UL * 6144 * 2);
  float*  outf = (float*)carve(16UL * 2048 * 4);
  float*  osum = (float*)carve(16UL * 2048 * 4);
  __bf16* A_cls = (__bf16*)carve(16UL * 2048 * 2);

  // ---- weight conversion to canonical bf16 [N,K] ----
  EW(conv_nt_kernel, 3072L * 640, W_ih[0], Wc_ih0[0], 3072, 513, 640);
  EW(conv_nt_kernel, 3072L * 640, W_ih[1], Wc_ih0[1], 3072, 513, 640);
  for (int j = 0; j < 4; ++j)
    EW(conv_nt_kernel, 3072L * 1024, W_hh[j], Wc_hh[j], 3072, 1024, 1024);
  EW(conv_nt_kernel, 3072L * 2048, W_ih[2], Wc_ih1[0], 3072, 2048, 2048);
  EW(conv_nt_kernel, 3072L * 2048, W_ih[3], Wc_ih1[1], 3072, 2048, 2048);
  EW(conv_t_kernel, 2048L * 2048, ENC_LIN_W, 2048, 0, WencL, 2048, 2048, 2048);
  EW(conv_t_kernel, 1024L * 2048, ATTN_W, 1024, 0,    WattE, 1024, 2048, 1024);
  EW(conv_t_kernel, 1024L * 2048, ATTN_W, 1024, 2048, WattD, 1024, 2048, 1024);
  for (int j = 0; j < 2; ++j)
    EW(conv_nt_kernel, 3072L * 4096, DW_ih[j], Wd_ih[j], 3072, 4096, 4096);
  for (int j = 0; j < 2; ++j)
    EW(conv_nt_kernel, 3072L * 1024, DW_hh[j], Wd_hh[j], 3072, 1024, 1024);
  EW(conv_t_kernel, 2048L * 6144, DEC_LIN_W, 2048, 0, WdecL, 2048, 6144, 2048);
  EW(conv_t_kernel, 1024L * 2048, CLS_W, 1000, 0, Wcls, 1000, 2048, 1024);

  // ---- encoder ----
  EW(pack_encin_kernel, 200L * 16 * 640, X, BORD, A_encin);
  gemm(stream, A_encin, 640, Wc_ih0[0], 640, giA, 3072, nullptr, 0, B_ih[0], 3200, 640, 3072, 3072, 0);
  gemm(stream, A_encin, 640, Wc_ih0[1], 640, giB, 3072, nullptr, 0, B_ih[1], 3200, 640, 3072, 3072, 0);
  gru_scan_kernel<<<1, 1024, 98304, stream>>>(giA, Wc_hh[0], B_hh[0], nullptr, xs1, 2048, 0,    hTf, 200, 0);
  gru_scan_kernel<<<1, 1024, 98304, stream>>>(giB, Wc_hh[1], B_hh[1], nullptr, xs1, 2048, 1024, hTb, 200, 1);
  gemm(stream, xs1, 2048, Wc_ih1[0], 2048, giA, 3072, nullptr, 0, B_ih[2], 3200, 2048, 3072, 3072, 0);
  gemm(stream, xs1, 2048, Wc_ih1[1], 2048, giB, 3072, nullptr, 0, B_ih[3], 3200, 2048, 3072, 3072, 0);
  gru_scan_kernel<<<1, 1024, 98304, stream>>>(giA, Wc_hh[2], B_hh[2], ys1f, nullptr, 0, 0, hTf, 200, 0);
  gru_scan_kernel<<<1, 1024, 98304, stream>>>(giB, Wc_hh[3], B_hh[3], ys1b, nullptr, 0, 0, hTb, 200, 1);
  ln_enc_kernel<<<3200, 256, 0, stream>>>(ys1f, ys1b, ENC_LN_G, ENC_LN_B, A_enc);
  gemm(stream, A_enc, 2048, WencL, 2048, enc_out, 2048, A_encout, 2048, ENC_LIN_B, 3200, 2048, 2048, 2048, 1);
  gemm(stream, A_encout, 2048, WattE, 2048, Epre, 1024, nullptr, 0, ATTN_B, 3200, 2048, 1024, 1024, 0);

  // ---- decoder init ----
  EW(f32_to_bf16_kernel, 16 * 1024, hTf, hbf_f, 16 * 1024);
  EW(f32_to_bf16_kernel, 16 * 1024, hTb, hbf_b, 16 * 1024);
  EW(zero_f32_kernel, 16 * 2048, osum, 16 * 2048);
  EW(emb_init_kernel, 16 * 2048, TGT, EMB, decin, gin, A_dl);

  // ---- decoder scan ----
  for (int t = 0; t < 40; ++t) {
    gemm(stream, decin, 2048, WattD, 2048, Dproj, 1024, nullptr, 0, nullptr, 16, 2048, 1024, 1024, 0);
    attn_step_kernel<<<16, 256, 0, stream>>>(Epre, Dproj, ATTN_V, ATTN_VB, enc_out, SST, SED, cbuf, gin, A_dl);
    gemm(stream, gin, 4096, Wd_ih[0], 4096, gidf, 3072, nullptr, 0, DB_ih[0], 16, 4096, 3072, 3072, 0);
    gemm(stream, gin, 4096, Wd_ih[1], 4096, gidb, 3072, nullptr, 0, DB_ih[1], 16, 4096, 3072, 3072, 0);
    gemm(stream, hbf_f, 1024, Wd_hh[0], 1024, ghdf, 3072, nullptr, 0, DB_hh[0], 16, 1024, 3072, 3072, 0);
    gemm(stream, hbf_b, 1024, Wd_hh[1], 1024, ghdb, 3072, nullptr, 0, DB_hh[1], 16, 1024, 3072, 3072, 0);
    EW(gru_combine_kernel, 16 * 1024, gidf, ghdf, hTf, hbf_f);
    EW(gru_combine_kernel, 16 * 1024, gidb, ghdb, hTb, hbf_b);
    ln_dec_kernel<<<16, 256, 0, stream>>>(hTf, hTb, DEC_LN_G, DEC_LN_B, A_dl);
    gemm(stream, A_dl, 6144, WdecL, 6144, outf, 2048, nullptr, 0, DEC_LIN_B, 16, 6144, 2048, 2048, 0);
    EW(dec_post_kernel, 16 * 2048, outf, osum, decin, gin, A_dl);
  }

  // ---- classifier on the mean (affine commutes with mean) ----
  EW(mean_kernel, 16 * 2048, osum, A_cls);
  gemm(stream, A_cls, 2048, Wcls, 2048, OUT, 1000, nullptr, 0, CLS_B, 16, 2048, 1000, 1024, 0);
}
